// MultiHeadAttention_30640296690175
// MI455X (gfx1250) — matrix operand order repaired and checked by reading, not compiled, not run on hardware
//
#include <hip/hip_runtime.h>
#include <hip/hip_bf16.h>

// ---------------------------------------------------------------------------
// MI455X (gfx1250) multi-head attention, bf16 WMMA, single-write attn output,
// async global->LDS staging (ASYNCcnt) with double-buffered GEMM K-loop.
// ---------------------------------------------------------------------------

typedef __bf16 bf16_t;
typedef __attribute__((ext_vector_type(16))) __bf16 v16bf;
typedef __attribute__((ext_vector_type(8)))  __bf16 v8bf;
typedef __attribute__((ext_vector_type(4)))  __bf16 v4bf;
typedef __attribute__((ext_vector_type(8)))  float  v8f;
typedef __attribute__((ext_vector_type(4)))  float  v4f;

#define B_   2
#define S_   2048
#define D_   1024
#define H_   16
#define DK_  64
#define M_   (B_ * S_)        // 4096 rows in projection GEMMs

// ---------------- CDNA5 async global->LDS copy helpers ----------------
// GLOBAL_LOAD_ASYNC_TO_LDS_B128: per-lane 16B copy, LDS addr in VDST VGPR,
// 64-bit global addr in VADDR pair; tracked by ASYNCcnt. The instruction
// offset applies to BOTH the LDS and global addresses (ISA 15.18.3), so one
// register pair covers a 32B copy with offset:16.
__device__ __forceinline__ void async_b128(void* lds, const void* g) {
    asm volatile("global_load_async_to_lds_b128 %0, %1, off"
                 :: "v"((unsigned)(unsigned long long)lds),
                    "v"((unsigned long long)g)
                 : "memory");
}
__device__ __forceinline__ void async_b128x2(void* lds, const void* g) {
    asm volatile("global_load_async_to_lds_b128 %0, %1, off\n\t"
                 "global_load_async_to_lds_b128 %0, %1, off offset:16"
                 :: "v"((unsigned)(unsigned long long)lds),
                    "v"((unsigned long long)g)
                 : "memory");
}
__device__ __forceinline__ void wait_async0() {
    asm volatile("s_wait_asynccnt 0x0" ::: "memory");
}

__device__ __forceinline__ v16bf make16(v8bf lo, v8bf hi) {
    return __builtin_shufflevector(lo, hi, 0,1,2,3,4,5,6,7,8,9,10,11,12,13,14,15);
}

// A-fragment (16x32 bf16, M x K): lane L holds row M = L&15;
// half 0 -> K = {kb+0..7, kb+16..23}, half 1 -> K = {kb+8..15, kb+24..31}
__device__ __forceinline__ v16bf frag_a(const bf16_t* tile, int row, int ld, int kb, int hf) {
    const bf16_t* p = tile + row * ld + kb + hf * 8;
    return make16(*(const v8bf*)p, *(const v8bf*)(p + 16));
}

// B-fragment (32x16 bf16, K x N): lane L holds col N = L&15;
// half 0 -> K = {kb+0..7, kb+16..23}, half 1 -> K = {kb+8..15, kb+24..31}
__device__ __forceinline__ v16bf frag_b(const bf16_t* tile, int col, int ld, int kb, int hf) {
    const bf16_t* p = tile + col * ld + kb + hf * 8;
    return make16(*(const v8bf*)p, *(const v8bf*)(p + 16));
}

__device__ __forceinline__ v8f wmma_bf16(v16bf a, v16bf b, v8f c) {
    return __builtin_amdgcn_wmma_f32_16x16x32_bf16(false, a, false, b, (short)0, c, false, false);
}

// ---------------------------------------------------------------------------
// f32 -> bf16 bulk convert (vectorized, n4 = n/4)
// ---------------------------------------------------------------------------
__global__ __launch_bounds__(256) void mha_cvt_bf16(const float* __restrict__ src,
                                                    bf16_t* __restrict__ dst, int n4) {
    int i = blockIdx.x * 256 + threadIdx.x;
    if (i < n4) {
        v4f x = *(const v4f*)(src + (size_t)i * 4);
        v4bf y;
        y[0] = (bf16_t)x[0]; y[1] = (bf16_t)x[1];
        y[2] = (bf16_t)x[2]; y[3] = (bf16_t)x[3];
        *(v4bf*)(dst + (size_t)i * 4) = y;
    }
}

// ---------------------------------------------------------------------------
// GEMM: out[m,n] = sum_k A[m,k] * W[n,k] + bias[n]
//   A: [4096,1024] bf16 row-major, W: [1024,1024] bf16 row-major (W^T via
//   B-fragment columns = W rows => contiguous LDS reads).
// mode 0..2: scatter bf16 out to head layout [B,H,S,DK] (q/k/v projections)
// mode 3   : f32 out row-major [4096,1024] (final output)
// Block: 128 threads (4 waves). Tile 64x64, K-step 32.
// Double-buffered LDS, filled by async global->LDS copies.
// ---------------------------------------------------------------------------
__global__ __launch_bounds__(128) void mha_gemm_bf16(const bf16_t* __restrict__ A,
                                                     const bf16_t* __restrict__ W,
                                                     const float*  __restrict__ bias,
                                                     bf16_t* __restrict__ outb,
                                                     float*  __restrict__ outf,
                                                     int mode) {
    __shared__ bf16_t At[2][64 * 32];   // 2 x 4 KB
    __shared__ bf16_t Wt[2][64 * 32];   // 2 x 4 KB

    const int tid  = threadIdx.x;
    const int wave = tid >> 5;
    const int lane = tid & 31;
    const int hf   = lane >> 4;
    const int lidx = lane & 15;

    const int m0 = blockIdx.x * 64;   // 4096 / 64 = 64 blocks
    const int n0 = blockIdx.y * 64;   // 1024 / 64 = 16 blocks

    v8f acc[4];
    #pragma unroll
    for (int nt = 0; nt < 4; nt++) { v8f z = {}; acc[nt] = z; }

    const int r = tid >> 1;           // 0..63
    const int c = (tid & 1) * 16;     // 0 or 16
    const bf16_t* Arow = A + (size_t)(m0 + r) * D_ + c;
    const bf16_t* Wrow = W + (size_t)(n0 + r) * D_ + c;

    // Prologue: fill buffer 0 asynchronously.
    async_b128x2(&At[0][r * 32 + c], Arow);
    async_b128x2(&Wt[0][r * 32 + c], Wrow);

    int buf = 0;
    for (int k0 = 0; k0 < D_; k0 += 32, buf ^= 1) {
        wait_async0();          // current buffer's async copies landed in LDS
        __syncthreads();        // all waves done with the buffer we overwrite next
        if (k0 + 32 < D_) {
            async_b128x2(&At[buf ^ 1][r * 32 + c], Arow + k0 + 32);
            async_b128x2(&Wt[buf ^ 1][r * 32 + c], Wrow + k0 + 32);
        }
        v16bf a = frag_a(&At[buf][0], 16 * wave + lidx, 32, 0, hf);
        #pragma unroll
        for (int nt = 0; nt < 4; nt++) {
            v16bf b = frag_b(&Wt[buf][0], nt * 16 + lidx, 32, 0, hf);
            acc[nt] = wmma_bf16(a, b, acc[nt]);
        }
    }

    // Epilogue. C-fragment: lane col = lidx, rows = hf*8 + v.
    #pragma unroll
    for (int nt = 0; nt < 4; nt++) {
        const int n  = n0 + nt * 16 + lidx;
        const float bv = bias[n];
        #pragma unroll
        for (int v = 0; v < 8; v++) {
            const int m = m0 + 16 * wave + hf * 8 + v;
            const float val = acc[nt][v] + bv;
            if (mode == 3) {
                outf[(size_t)m * D_ + n] = val;
            } else {
                const int b = m >> 11, s = m & (S_ - 1);
                const int h = n >> 6,  d = n & (DK_ - 1);
                outb[(((size_t)(b * H_ + h)) * S_ + s) * DK_ + d] = (bf16_t)val;
            }
        }
    }
}

// ---------------------------------------------------------------------------
// Attention: per block = one (b,h) and one 64-row query chunk.
// Block 128 threads / 4 waves; wave w owns query rows 16w..16w+15 (stats are
// wave-local registers, reduced with 16-lane shuffles -> no barriers).
// Pass 1: online softmax stats (m, l) over causal key tiles.
// Pass 2: recompute scores, write normalized p to attn ONCE (536 MB floor),
//         accumulate ctx += p @ V via WMMA (p transposed through LDS).
// q/k tiles staged with async global->LDS copies.
// ---------------------------------------------------------------------------
__global__ __launch_bounds__(128) void mha_attn(const bf16_t* __restrict__ qh,
                                                const bf16_t* __restrict__ kh,
                                                const bf16_t* __restrict__ vh,
                                                float* __restrict__ attn,
                                                bf16_t* __restrict__ ctx) {
    constexpr int LDT = 72;                    // padded stride (bank-friendly)
    __shared__ bf16_t qt[64 * LDT];            // 9 KB
    __shared__ bf16_t kt[64 * LDT];            // 9 KB
    __shared__ bf16_t vtT[64 * LDT];           // 9 KB (V transposed: [d][key])
    __shared__ bf16_t pt[4][16 * LDT];         // 9 KB (per-wave p tiles)

    const int bh = blockIdx.x;                 // b*16 + h
    const int qb = blockIdx.y;                 // query tile, 2048/64 = 32
    const int b  = bh >> 4, h = bh & 15;
    const int tid  = threadIdx.x;
    const int wave = tid >> 5;
    const int lane = tid & 31;
    const int hf   = lane >> 4;
    const int lidx = lane & 15;
    const int q0   = qb * 64;
    const int myrow0 = 16 * wave;
    const float scale = 0.125f;                // 1/sqrt(64)

    const bf16_t* qbase = qh + (size_t)bh * S_ * DK_;
    const bf16_t* kbase = kh + (size_t)bh * S_ * DK_;
    const bf16_t* vbase = vh + (size_t)bh * S_ * DK_;
    float* attnb = attn + (size_t)bh * S_ * S_ + (size_t)q0 * S_;

    // Async-load q tile [64 x 64] once (waited together with first k tile).
    #pragma unroll
    for (int i = tid; i < 512; i += 128) {
        int rr = i >> 3, cc = (i & 7) * 8;
        async_b128(&qt[rr * LDT + cc], qbase + (size_t)(q0 + rr) * DK_ + cc);
    }

    const int diagTile = qb;                   // last key tile with any valid key

    float mrow[8], lrow[8];
    #pragma unroll
    for (int v = 0; v < 8; v++) { mrow[v] = -1e30f; lrow[v] = 0.0f; }

    v8f cacc[4];
    #pragma unroll
    for (int dt = 0; dt < 4; dt++) { v8f z = {}; cacc[dt] = z; }

    // ---------------- Pass 1: softmax stats ----------------
    for (int ktile = 0; ktile <= diagTile; ktile++) {
        const int k0 = ktile * 64;
        __syncthreads();                       // readers of previous kt done
        #pragma unroll
        for (int i = tid; i < 512; i += 128) {
            int rr = i >> 3, cc = (i & 7) * 8;
            async_b128(&kt[rr * LDT + cc], kbase + (size_t)(k0 + rr) * DK_ + cc);
        }
        wait_async0();
        __syncthreads();

        v8f sf[4];
        #pragma unroll
        for (int nt = 0; nt < 4; nt++) { v8f z = {}; sf[nt] = z; }
        #pragma unroll
        for (int kk = 0; kk < 64; kk += 32) {
            v16bf a = frag_a(qt, myrow0 + lidx, LDT, kk, hf);
            #pragma unroll
            for (int nt = 0; nt < 4; nt++) {
                v16bf bb = frag_b(kt, nt * 16 + lidx, LDT, kk, hf);
                sf[nt] = wmma_bf16(a, bb, sf[nt]);
            }
        }

        float tmax[8];
        #pragma unroll
        for (int v = 0; v < 8; v++) tmax[v] = -1e30f;
        #pragma unroll
        for (int nt = 0; nt < 4; nt++) {
            const int ki = k0 + nt * 16 + lidx;
            #pragma unroll
            for (int v = 0; v < 8; v++) {
                const int qi = q0 + myrow0 + hf * 8 + v;
                float s = sf[nt][v] * scale;
                if (ki > qi) s = -1e30f;   // causal mask
                sf[nt][v] = s;
                tmax[v] = fmaxf(tmax[v], s);
            }
        }
        #pragma unroll
        for (int off = 1; off < 16; off <<= 1) {
            #pragma unroll
            for (int v = 0; v < 8; v++)
                tmax[v] = fmaxf(tmax[v], __shfl_xor(tmax[v], off, 32));
        }
        #pragma unroll
        for (int v = 0; v < 8; v++) {
            const float mn = fmaxf(mrow[v], tmax[v]);
            float ssum = 0.0f;
            #pragma unroll
            for (int nt = 0; nt < 4; nt++) ssum += __expf(sf[nt][v] - mn);
            #pragma unroll
            for (int off = 1; off < 16; off <<= 1) ssum += __shfl_xor(ssum, off, 32);
            lrow[v] = lrow[v] * __expf(mrow[v] - mn) + ssum;
            mrow[v] = mn;
        }
    }

    float rl[8];
    #pragma unroll
    for (int v = 0; v < 8; v++) rl[v] = 1.0f / lrow[v];

    // ---------------- Pass 2: write attn once + ctx = p @ V ----------------
    for (int ktile = 0; ktile < S_ / 64; ktile++) {
        const int k0 = ktile * 64;
        if (ktile > diagTile) {
            // fully masked tile: zero-fill attn (uniform branch; no barriers here)
            for (int i = tid; i < 1024; i += 128) {
                int rr = i >> 4, cc = (i & 15) << 2;
                v4f z = {};
                *(v4f*)(attnb + (size_t)rr * S_ + k0 + cc) = z;
            }
            continue;
        }
        __syncthreads();
        #pragma unroll
        for (int i = tid; i < 512; i += 128) {
            int rr = i >> 3, cc = (i & 7) * 8;
            async_b128(&kt[rr * LDT + cc], kbase + (size_t)(k0 + rr) * DK_ + cc);
            // V transposed into LDS: vtT[d][key] (element shuffle, regular path)
            v8bf vv = *(const v8bf*)(vbase + (size_t)(k0 + rr) * DK_ + cc);
            #pragma unroll
            for (int j = 0; j < 8; j++) vtT[(cc + j) * LDT + rr] = vv[j];
        }
        wait_async0();
        __syncthreads();

        v8f sf[4];
        #pragma unroll
        for (int nt = 0; nt < 4; nt++) { v8f z = {}; sf[nt] = z; }
        #pragma unroll
        for (int kk = 0; kk < 64; kk += 32) {
            v16bf a = frag_a(qt, myrow0 + lidx, LDT, kk, hf);
            #pragma unroll
            for (int nt = 0; nt < 4; nt++) {
                v16bf bb = frag_b(kt, nt * 16 + lidx, LDT, kk, hf);
                sf[nt] = wmma_bf16(a, bb, sf[nt]);
            }
        }

        // p = exp(s*scale - m) / l ; single global write of attn; stage bf16 p.
        #pragma unroll
        for (int nt = 0; nt < 4; nt++) {
            const int ki = k0 + nt * 16 + lidx;
            #pragma unroll
            for (int v = 0; v < 8; v++) {
                const int rrow = myrow0 + hf * 8 + v;
                const int qi = q0 + rrow;
                float p = __expf(sf[nt][v] * scale - mrow[v]) * rl[v];
                if (ki > qi) p = 0.0f;
                attnb[(size_t)rrow * S_ + ki] = p;
                pt[wave][(hf * 8 + v) * LDT + nt * 16 + lidx] = (bf16_t)p;
            }
        }

        // ctx += p @ V  (A = pt (16x64), B = vtT columns = d-dim)
        #pragma unroll
        for (int kk = 0; kk < 64; kk += 32) {
            v16bf a = frag_a(&pt[wave][0], lidx, LDT, kk, hf);
            #pragma unroll
            for (int dt = 0; dt < 4; dt++) {
                v16bf bb = frag_b(vtT, dt * 16 + lidx, LDT, kk, hf);
                cacc[dt] = wmma_bf16(a, bb, cacc[dt]);
            }
        }
    }

    // Store context in [B,S,D] layout (heads re-interleaved) as bf16.
    #pragma unroll
    for (int dt = 0; dt < 4; dt++) {
        const int d = dt * 16 + lidx;
        #pragma unroll
        for (int v = 0; v < 8; v++) {
            const int qi = q0 + myrow0 + hf * 8 + v;
            ctx[((size_t)b * S_ + qi) * D_ + h * DK_ + d] = (bf16_t)cacc[dt][v];
        }
    }
}

// ---------------------------------------------------------------------------
// Host-side launch. Workspace layout (bytes), total 64 MB:
//   [ 0, 8M)  Qb     bf16 [4096,1024]
//   [ 8,16M)  Kb
//   [16,24M)  Vb
//   [24,26M)  WQb    bf16 [1024,1024]
//   [26,28M)  WKb
//   [28,30M)  WVb
//   [30,32M)  WOb
//   [32,40M)  q_h    bf16 [2,16,2048,64]
//   [40,48M)  k_h
//   [48,56M)  v_h
//   [56,64M)  ctx    bf16 [2,2048,1024]
// ---------------------------------------------------------------------------
extern "C" void kernel_launch(void* const* d_in, const int* in_sizes, int n_in,
                              void* d_out, int out_size, void* d_ws, size_t ws_size,
                              hipStream_t stream) {
    (void)in_sizes; (void)n_in; (void)out_size; (void)ws_size;

    const float* Q   = (const float*)d_in[0];
    const float* K   = (const float*)d_in[1];
    const float* V   = (const float*)d_in[2];
    /* d_in[3] = causal mask (implicit in kernel) */
    const float* W_Q = (const float*)d_in[4];
    const float* b_Q = (const float*)d_in[5];
    const float* W_K = (const float*)d_in[6];
    const float* b_K = (const float*)d_in[7];
    const float* W_V = (const float*)d_in[8];
    const float* b_V = (const float*)d_in[9];
    const float* W_O = (const float*)d_in[10];
    const float* b_O = (const float*)d_in[11];

    char* ws = (char*)d_ws;
    const size_t MB = 1ull << 20;
    bf16_t* Qb  = (bf16_t*)(ws + 0 * MB);
    bf16_t* Kb  = (bf16_t*)(ws + 8 * MB);
    bf16_t* Vb  = (bf16_t*)(ws + 16 * MB);
    bf16_t* WQb = (bf16_t*)(ws + 24 * MB);
    bf16_t* WKb = (bf16_t*)(ws + 26 * MB);
    bf16_t* WVb = (bf16_t*)(ws + 28 * MB);
    bf16_t* WOb = (bf16_t*)(ws + 30 * MB);
    bf16_t* q_h = (bf16_t*)(ws + 32 * MB);
    bf16_t* k_h = (bf16_t*)(ws + 40 * MB);
    bf16_t* v_h = (bf16_t*)(ws + 48 * MB);
    bf16_t* ctx = (bf16_t*)(ws + 56 * MB);

    float* out  = (float*)d_out;                       // [2,2048,1024]
    float* attn = out + (size_t)B_ * S_ * D_;          // [2,16,2048,2048]

    // f32 -> bf16 conversions
    const int nX4 = (M_ * D_) / 4;        // 1,048,576
    const int nW4 = (D_ * D_) / 4;        // 262,144
    mha_cvt_bf16<<<nX4 / 256, 256, 0, stream>>>(Q, Qb, nX4);
    mha_cvt_bf16<<<nX4 / 256, 256, 0, stream>>>(K, Kb, nX4);
    mha_cvt_bf16<<<nX4 / 256, 256, 0, stream>>>(V, Vb, nX4);
    mha_cvt_bf16<<<nW4 / 256, 256, 0, stream>>>(W_Q, WQb, nW4);
    mha_cvt_bf16<<<nW4 / 256, 256, 0, stream>>>(W_K, WKb, nW4);
    mha_cvt_bf16<<<nW4 / 256, 256, 0, stream>>>(W_V, WVb, nW4);
    mha_cvt_bf16<<<nW4 / 256, 256, 0, stream>>>(W_O, WOb, nW4);

    // Projections (scatter into [B,H,S,DK])
    dim3 ggrid(M_ / 64, D_ / 64);
    mha_gemm_bf16<<<ggrid, 128, 0, stream>>>(Qb, WQb, b_Q, q_h, nullptr, 0);
    mha_gemm_bf16<<<ggrid, 128, 0, stream>>>(Kb, WKb, b_K, k_h, nullptr, 1);
    mha_gemm_bf16<<<ggrid, 128, 0, stream>>>(Vb, WVb, b_V, v_h, nullptr, 2);

    // Attention (writes attn exactly once, produces ctx)
    mha_attn<<<dim3(B_ * H_, S_ / 64), 128, 0, stream>>>(q_h, k_h, v_h, attn, ctx);

    // Output projection (f32 out)
    mha_gemm_bf16<<<ggrid, 128, 0, stream>>>(ctx, WOb, b_O, nullptr, out, 3);
}
